// InteractionBlock_49417893708203
// MI455X (gfx1250) — matrix operand order repaired and checked by reading, not compiled, not run on hardware
//
#include <hip/hip_runtime.h>
#include <cstdint>

// ---------------------------------------------------------------------------
// Types / helpers
// ---------------------------------------------------------------------------
typedef __attribute__((ext_vector_type(16))) __bf16 v16bf;
typedef __attribute__((ext_vector_type(8)))  float  v8f;

#define DEV __device__ __forceinline__

DEV v8f wmma_bf16(v16bf a, v16bf b, v8f c) {
  // D = A(16x32) * B(32x16) + C(16x16 f32)
  return __builtin_amdgcn_wmma_f32_16x16x32_bf16(false, a, false, b, (short)0, c,
                                                 false, false);
}

// Branch-free shifted softplus: softplus(beta*x) - log(2).
// softplus(z) = max(z,0) + log(1 + exp(-|z|)); uses HW v_exp_f32 / v_log_f32.
DEV float ssp_f(float x, float beta) {
  float z = beta * x;
  float e = __expf(-__builtin_fabsf(z));
  float sp = __builtin_fmaxf(z, 0.f) + __logf(1.f + e);
  return sp - 0.69314718056f;
}

// Fast sigmoid: HW exp + HW rcp (v_rcp_f32), no precise-div expansion.
DEV float sigmoid_f(float x) {
  return __builtin_amdgcn_rcpf(1.f + __expf(-x));
}

DEV void atomic_add_f32(float* p, float v) {
  __hip_atomic_fetch_add(p, v, __ATOMIC_RELAXED, __HIP_MEMORY_SCOPE_AGENT);
}

// A fragment (16x32, bf16) built from a row-major f32 row pointer (&A[m*lda]).
// Lane mapping (ISA 16-bit A 16x32): lanes 0-15 -> K {0..7, 16..23},
// lanes 16-31 -> K {8..15, 24..31}, m = lane & 15.
DEV v16bf a_frag_f32(const float* __restrict__ row, int k0, int lane) {
  int kh = (lane >> 4) & 1;
  const float* p0 = row + k0 + kh * 8;
  const float* p1 = row + k0 + 16 + kh * 8;
  v16bf a;
#pragma unroll
  for (int j = 0; j < 8; ++j) a[j] = (__bf16)p0[j];
#pragma unroll
  for (int j = 0; j < 8; ++j) a[8 + j] = (__bf16)p1[j];
  return a;
}

// Bounded variant: zero-pad K >= ksrc (used only for the K=32..63 tail of
// edge_attr, K=50 padded to 64).
DEV v16bf a_frag_f32_b(const float* __restrict__ row, int k0, int ksrc, int lane) {
  int kh = (lane >> 4) & 1;
  v16bf a;
#pragma unroll
  for (int j = 0; j < 8; ++j) {
    int k = k0 + kh * 8 + j;
    a[j] = (__bf16)((k < ksrc) ? row[k] : 0.f);
  }
#pragma unroll
  for (int j = 0; j < 8; ++j) {
    int k = k0 + 16 + kh * 8 + j;
    a[8 + j] = (__bf16)((k < ksrc) ? row[k] : 0.f);
  }
  return a;
}

// A fragment from LDS bf16 row-major (row = &sA[m*ld]).
DEV v16bf a_frag_lds(const __bf16* __restrict__ row, int k0, int lane) {
  int kh = (lane >> 4) & 1;
  const __bf16* p0 = row + k0 + kh * 8;
  const __bf16* p1 = row + k0 + 16 + kh * 8;
  v16bf a;
#pragma unroll
  for (int j = 0; j < 8; ++j) a[j] = p0[j];
#pragma unroll
  for (int j = 0; j < 8; ++j) a[8 + j] = p1[j];
  return a;
}

// Packed B fragment: pack layout is [(kt*NT+nt)*32 + lane][16] contiguous bf16,
// i.e. one 32-byte load per lane.
DEV v16bf b_frag(const __bf16* __restrict__ Bp, int kt, int nt, int NT, int lane) {
  const __bf16* p = Bp + ((size_t)((kt * NT + nt) * 32 + lane) << 4);
  v16bf b;
#pragma unroll
  for (int j = 0; j < 16; ++j) b[j] = p[j];
  return b;
}

// ---------------------------------------------------------------------------
// Weight packing: B (K x N, row-major f32, ld=N) -> WMMA lane-fragment order.
// Assumed dense bf16 B layout (32x16 tile): lanes 0-15 hold K=kt*32+{0..7,16..23} for
// column n=lane; lanes 16-31 hold K=kt*32+{8..15,24..31}.
// ---------------------------------------------------------------------------
__global__ void k_pack_b(const float* __restrict__ src, int Ksrc, int Ncols,
                         __bf16* __restrict__ dst, int KT, int NT) {
  int idx = blockIdx.x * blockDim.x + threadIdx.x;
  int total = KT * NT * 512;
  if (idx >= total) return;
  int j = idx & 15;
  int lane = (idx >> 4) & 31;
  int tile = idx >> 9;
  int nt = tile % NT, kt = tile / NT;
  int n = nt * 16 + (lane & 15);
  int k = kt * 32 + ((lane >> 4) << 3) + (j & 7) + ((j >> 3) << 4);
  float v = (k < Ksrc) ? src[(size_t)k * Ncols + n] : 0.f;
  dst[idx] = (__bf16)v;
}

__global__ void k_zero(float* __restrict__ p, long n) {
  long i = (long)blockIdx.x * blockDim.x + threadIdx.x;
  long stride = (long)gridDim.x * blockDim.x;
  for (; i < n; i += stride) p[i] = 0.f;
}

// ---------------------------------------------------------------------------
// Generic WMMA GEMM: C[M,NCOLS] = A[M,K](f32,row-major) @ Bpacked + bias.
// One wave per 16-row tile; wave-uniform guard keeps EXEC all-ones for WMMA.
// ---------------------------------------------------------------------------
template <int K, int NCOLS>
__global__ void k_gemm(const float* __restrict__ A, const __bf16* __restrict__ Bp,
                       const float* __restrict__ bias, float* __restrict__ C,
                       int Mtiles) {
  constexpr int KT = K / 32;
  constexpr int NT = NCOLS / 16;
  int lane = threadIdx.x & 31;
  int tile = (blockIdx.x * blockDim.x + threadIdx.x) >> 5;
  if (tile >= Mtiles) return;
  const float* Arow = A + ((size_t)tile * 16 + (lane & 15)) * K;
  v16bf af[KT];
#pragma unroll
  for (int kt = 0; kt < KT; ++kt) af[kt] = a_frag_f32(Arow, kt * 32, lane);
  int n0 = lane & 15;
  int kh = lane >> 4;
#pragma unroll
  for (int nt = 0; nt < NT; ++nt) {
    v8f acc = {0.f, 0.f, 0.f, 0.f, 0.f, 0.f, 0.f, 0.f};
#pragma unroll
    for (int kt = 0; kt < KT; ++kt)
      acc = wmma_bf16(af[kt], b_frag(Bp, kt, nt, NT, lane), acc);
    int n = nt * 16 + n0;
    float b = bias[n];
    float* crow = C + ((size_t)tile * 16 + kh * 8) * NCOLS + n;
#pragma unroll
    for (int r = 0; r < 8; ++r) crow[(size_t)r * NCOLS] = acc[r] + b;
  }
}

// ---------------------------------------------------------------------------
// BatchNorm (training mode) support
// ---------------------------------------------------------------------------
__global__ void k_colstats(const float* __restrict__ H, int nf, int nrows,
                           int rowsPerBlock, float* __restrict__ sum,
                           float* __restrict__ sumsq) {
  int f = threadIdx.x;  // blockDim.x == nf
  int r0 = blockIdx.x * rowsPerBlock;
  int r1 = r0 + rowsPerBlock;
  if (r1 > nrows) r1 = nrows;
  float s = 0.f, ss = 0.f;
  for (int r = r0; r < r1; ++r) {
    float v = H[(size_t)r * nf + f];
    s += v;
    ss += v * v;
  }
  atomic_add_f32(&sum[f], s);
  atomic_add_f32(&sumsq[f], ss);
}

__global__ void k_bn_finalize(const float* __restrict__ sum,
                              const float* __restrict__ sumsq,
                              const float* __restrict__ g,
                              const float* __restrict__ b, float invN, int nf,
                              float* __restrict__ scale,
                              float* __restrict__ shift) {
  int f = blockIdx.x * blockDim.x + threadIdx.x;
  if (f >= nf) return;
  float mean = sum[f] * invN;
  float var = sumsq[f] * invN - mean * mean;
  float sc = g[f] * rsqrtf(var + 1e-5f);
  scale[f] = sc;
  shift[f] = b[f] - mean * sc;
}

__global__ void k_bn_leaky(float* __restrict__ H, const float* __restrict__ scale,
                           const float* __restrict__ shift, int nf, long total) {
  long i = (long)blockIdx.x * blockDim.x + threadIdx.x;
  if (i >= total) return;
  int f = (int)(i & (nf - 1));  // nf is a power of two (128 or 64)
  float v = H[i] * scale[f] + shift[f];
  H[i] = v > 0.f ? v : 0.2f * v;
}

// ---------------------------------------------------------------------------
// Per-edge scalar weight: sigmoid(relu(l*w1+b1)@w2 + b2) * cos-cutoff * mask
// ---------------------------------------------------------------------------
__global__ void k_combined(const float* __restrict__ elen,
                           const float* __restrict__ w1, const float* __restrict__ b1,
                           const float* __restrict__ w2, const float* __restrict__ b2,
                           float* __restrict__ out, int E) {
  int e = blockIdx.x * blockDim.x + threadIdx.x;
  if (e >= E) return;
  float l = elen[e];
  float s = b2[0];
#pragma unroll
  for (int j = 0; j < 32; ++j) {
    float t = l * w1[j] + b1[j];
    s += (t > 0.f ? t : 0.f) * w2[j];
  }
  float lw = sigmoid_f(s);
  float Cc = 0.5f * (__cosf(l * 0.31415926535f) + 1.f);  // pi/CUTOFF
  float m = (l <= 10.f && l >= 0.f) ? 1.f : 0.f;
  out[e] = lw * Cc * m;
}

// ---------------------------------------------------------------------------
// Fused per-edge pipeline (one wave per 16 edges, both conv pathways):
//   W = (ssp(attr@w1 + b1) @ w2 + b2) * combined   (never leaves LDS)
//   msg = h[src] * W ;  agg[dst] += msg  (f32 hw atomics)
// h[src] rows are prefetched (global_prefetch_b8) at wave start so the random
// gather is L1/L2-resident by the time the filter MLP finishes.
// ---------------------------------------------------------------------------
template <int NF>
DEV void conv_path(int lane, int e0, const float* __restrict__ comb,
                   const v16bf& ea0, const v16bf& ea1,
                   const __bf16* __restrict__ w1p, const float* __restrict__ b1,
                   const float* __restrict__ pbeta,
                   const __bf16* __restrict__ w2p, const float* __restrict__ b2,
                   const float* __restrict__ h, float* __restrict__ agg,
                   const int* __restrict__ sS, const int* __restrict__ sD,
                   __bf16* __restrict__ sA, float* __restrict__ sW,
                   float* __restrict__ sC) {
  constexpr int NT = NF / 16;
  constexpr int KT = NF / 32;
  if (lane < 16) sC[lane] = comb[e0 + lane];
  float beta = pbeta[0];
  int n0 = lane & 15;
  int row0 = (lane >> 4) * 8;
  // GEMM1 (K=64, edge_attr padded) + bias + SSP -> sA (bf16, row-major 16xNF)
#pragma unroll
  for (int nt = 0; nt < NT; ++nt) {
    v8f acc = {0.f, 0.f, 0.f, 0.f, 0.f, 0.f, 0.f, 0.f};
    acc = wmma_bf16(ea0, b_frag(w1p, 0, nt, NT, lane), acc);
    acc = wmma_bf16(ea1, b_frag(w1p, 1, nt, NT, lane), acc);
    float bb = b1[nt * 16 + n0];
#pragma unroll
    for (int r = 0; r < 8; ++r)
      sA[(row0 + r) * NF + nt * 16 + n0] = (__bf16)ssp_f(acc[r] + bb, beta);
  }
  // GEMM2 (K=NF) + bias, scaled by per-edge combined -> sW (f32)
  const __bf16* arow = sA + n0 * NF;  // m = lane & 15
#pragma unroll
  for (int nt = 0; nt < NT; ++nt) {
    v8f acc = {0.f, 0.f, 0.f, 0.f, 0.f, 0.f, 0.f, 0.f};
#pragma unroll
    for (int kt = 0; kt < KT; ++kt)
      acc = wmma_bf16(a_frag_lds(arow, kt * 32, lane),
                      b_frag(w2p, kt, nt, NT, lane), acc);
    float bb = b2[nt * 16 + n0];
#pragma unroll
    for (int r = 0; r < 8; ++r)
      sW[(row0 + r) * NF + nt * 16 + n0] = (acc[r] + bb) * sC[row0 + r];
  }
  // gather h[src], multiply, scatter-add to agg[dst]
#pragma unroll 1
  for (int e = 0; e < 16; ++e) {
    int s = sS[e], d = sD[e];
    const float* hrow = h + (size_t)s * NF;
    float* grow = agg + (size_t)d * NF;
#pragma unroll
    for (int f = lane; f < NF; f += 32)
      atomic_add_f32(&grow[f], hrow[f] * sW[e * NF + f]);
  }
}

__global__ void __launch_bounds__(128) k_edge_conv(
    const float* __restrict__ eattr, const int* __restrict__ src,
    const int* __restrict__ dst, const float* __restrict__ comb1,
    const float* __restrict__ comb2, const float* __restrict__ h1,
    const float* __restrict__ h2, float* __restrict__ agg1,
    float* __restrict__ agg2, const __bf16* __restrict__ w1p1,
    const float* __restrict__ b1c1, const float* __restrict__ beta1,
    const __bf16* __restrict__ w2p1, const float* __restrict__ b2c1,
    const __bf16* __restrict__ w1p2, const float* __restrict__ b1c2,
    const float* __restrict__ beta2, const __bf16* __restrict__ w2p2,
    const float* __restrict__ b2c2, int Etiles) {
  __shared__ __bf16 sA[4][16 * 128];
  __shared__ float sW[4][16 * 128];
  __shared__ int sS[4][16];
  __shared__ int sD[4][16];
  __shared__ float sC[4][16];
  int lane = threadIdx.x & 31;
  int wv = threadIdx.x >> 5;
  int tile = blockIdx.x * 4 + wv;
  if (tile >= Etiles) return;  // wave-uniform
  int e0 = tile * 16;
  // All 32 lanes read a src index (lanes 16-31 duplicate rows 0-15) so every
  // lane can prefetch one 128B chunk of the h-rows it will gather later.
  int si = src[e0 + (lane & 15)];
  if (lane < 16) {
    sS[wv][lane] = si;
    sD[wv][lane] = dst[e0 + lane];
  }
  {
    // conv1 rows are 512B (4 x 128B); conv2 rows are 256B (2 x 128B).
    const char* r1 = (const char*)(h1 + (size_t)si * 128) + (lane >> 4) * 256;
    __builtin_prefetch(r1, 0, 3);
    __builtin_prefetch(r1 + 128, 0, 3);
    const char* r2 = (const char*)(h2 + (size_t)si * 64) + (lane >> 4) * 128;
    __builtin_prefetch(r2, 0, 3);
  }
  const float* arow = eattr + (size_t)(e0 + (lane & 15)) * 50;
  v16bf ea0 = a_frag_f32(arow, 0, lane);         // K 0..31 all < 50: no bounds
  v16bf ea1 = a_frag_f32_b(arow, 32, 50, lane);  // K 32..63: pad >= 50
  conv_path<128>(lane, e0, comb1, ea0, ea1, w1p1, b1c1, beta1, w2p1, b2c1, h1,
                 agg1, sS[wv], sD[wv], sA[wv], sW[wv], sC[wv]);
  conv_path<64>(lane, e0, comb2, ea0, ea1, w1p2, b1c2, beta2, w2p2, b2c2, h2,
                agg2, sS[wv], sD[wv], sA[wv], sW[wv], sC[wv]);
}

// ---------------------------------------------------------------------------
// Tail: BN2-apply + SSP into concat buffer; final linear; attention; output
// ---------------------------------------------------------------------------
__global__ void k_concat_ssp(const float* __restrict__ o1, const float* __restrict__ o2,
                             const float* __restrict__ sc1, const float* __restrict__ sh1,
                             const float* __restrict__ sc2, const float* __restrict__ sh2,
                             const float* __restrict__ pbeta,
                             float* __restrict__ xcat, long total) {
  long i = (long)blockIdx.x * blockDim.x + threadIdx.x;
  if (i >= total) return;
  int c = (int)(i & 255);
  long r = i >> 8;
  float beta = pbeta[0];
  float v;
  if (c < 128) {
    v = o1[r * 128 + c] * sc1[c] + sh1[c];
  } else {
    int f = c - 128;
    v = o2[r * 128 + f] * sc2[f] + sh2[f];
  }
  xcat[i] = ssp_f(v, beta);
}

__global__ void k_attn(const float* __restrict__ T /*[N,64] pre-relu*/,
                       const float* __restrict__ w2, const float* __restrict__ b2,
                       float* __restrict__ attnv, int N) {
  int n = blockIdx.x * blockDim.x + threadIdx.x;
  if (n >= N) return;
  float s = b2[0];
#pragma unroll
  for (int j = 0; j < 64; ++j) {
    float t = T[(size_t)n * 64 + j];
    s += (t > 0.f ? t : 0.f) * w2[j];
  }
  attnv[n] = sigmoid_f(s);
}

__global__ void k_final(const float* __restrict__ xc, const float* __restrict__ attnv,
                        float* __restrict__ out, long total) {
  long i = (long)blockIdx.x * blockDim.x + threadIdx.x;
  if (i >= total) return;
  out[i] = xc[i] * attnv[i >> 7];
}

// ---------------------------------------------------------------------------
// Host launcher
// ---------------------------------------------------------------------------
extern "C" void kernel_launch(void* const* d_in, const int* in_sizes, int n_in,
                              void* d_out, int out_size, void* d_ws, size_t ws_size,
                              hipStream_t stream) {
  constexpr int N = 50000, E = 800000;
  constexpr int MT_N = N / 16;  // 3125 node row-tiles
  constexpr int ET = E / 16;    // 50000 edge tiles

  // Input mapping: setup_inputs() dict insertion order, params flattened in
  // insertion order (conv1, conv2, act_beta, lin_w/b, attn_w1/b1, attn_w2/b2).
  const float* x = (const float*)d_in[0];
  const int* eidx = (const int*)d_in[1];
  const int* src = eidx;
  const int* dst = eidx + E;
  const float* elen = (const float*)d_in[2];
  const float* eattr = (const float*)d_in[3];
  auto F = [&](int i) { return (const float*)d_in[i]; };
  // conv block leaf offsets (relative): lin1_w,lin1_b,bn1_g,bn1_b,lin2_w,lin2_b,
  // bn2_g,bn2_b,mlp_w1,mlp_b1,mlp_beta,mlp_w2,mlp_b2,dw_w1,dw_b1,dw_w2,dw_b2
  const int C1 = 4, C2 = 21;
  const int L_LIN1W = 0, L_LIN1B = 1, L_BN1G = 2, L_BN1B = 3, L_LIN2W = 4,
            L_LIN2B = 5, L_BN2G = 6, L_BN2B = 7, L_MLPW1 = 8, L_MLPB1 = 9,
            L_MLPBETA = 10, L_MLPW2 = 11, L_MLPB2 = 12, L_DWW1 = 13,
            L_DWB1 = 14, L_DWW2 = 15, L_DWB2 = 16;
  const int I_ACTBETA = 38, I_LINW = 39, I_LINB = 40, I_ATTNW1 = 41,
            I_ATTNB1 = 42, I_ATTNW2 = 43, I_ATTNB2 = 44;
  (void)in_sizes; (void)n_in; (void)out_size; (void)ws_size;

  // workspace carve
  char* w = (char*)d_ws;
  size_t off = 0;
  auto alloc = [&](size_t bytes) -> void* {
    void* p = w + off;
    off = (off + bytes + 255) & ~(size_t)255;
    return p;
  };
  __bf16* lin1p1 = (__bf16*)alloc(128 * 128 * 2);
  __bf16* lin1p2 = (__bf16*)alloc(128 * 64 * 2);
  __bf16* lin2p1 = (__bf16*)alloc(128 * 128 * 2);
  __bf16* lin2p2 = (__bf16*)alloc(64 * 128 * 2);
  __bf16* w1p1 = (__bf16*)alloc(64 * 128 * 2);
  __bf16* w1p2 = (__bf16*)alloc(64 * 64 * 2);
  __bf16* w2p1 = (__bf16*)alloc(128 * 128 * 2);
  __bf16* w2p2 = (__bf16*)alloc(64 * 64 * 2);
  __bf16* linp = (__bf16*)alloc(256 * 128 * 2);
  __bf16* attn1p = (__bf16*)alloc(128 * 64 * 2);
  float* comb1 = (float*)alloc((size_t)E * 4);
  float* comb2 = (float*)alloc((size_t)E * 4);
  float* h1 = (float*)alloc((size_t)N * 128 * 4);
  float* h2 = (float*)alloc((size_t)N * 64 * 4);
  float* agg1 = (float*)alloc((size_t)N * 128 * 4);
  float* agg2 = (float*)alloc((size_t)N * 64 * 4);
  float* o1 = (float*)alloc((size_t)N * 128 * 4);
  float* o2 = (float*)alloc((size_t)N * 128 * 4);
  float* xcat = (float*)alloc((size_t)N * 256 * 4);
  float* xc = (float*)alloc((size_t)N * 128 * 4);
  float* tbuf = (float*)alloc((size_t)N * 64 * 4);
  float* attnv = (float*)alloc((size_t)N * 4);
  float* st = (float*)alloc(4096 * 4);
  // stats layout (floats)
  float* sum_h1 = st + 0;     float* sq_h1 = st + 128;
  float* sum_h2 = st + 256;   float* sq_h2 = st + 320;
  float* sum_o1 = st + 384;   float* sq_o1 = st + 512;
  float* sum_o2 = st + 640;   float* sq_o2 = st + 768;
  float* scale_h1 = st + 896; float* shift_h1 = st + 1024;
  float* scale_h2 = st + 1152; float* shift_h2 = st + 1216;
  float* scale_o1 = st + 1280; float* shift_o1 = st + 1408;
  float* scale_o2 = st + 1536; float* shift_o2 = st + 1664;

  auto cdiv = [](long a, long b) { return (int)((a + b - 1) / b); };
  const float invN = 1.f / (float)N;

  // zero accumulators + stats sums
  k_zero<<<2048, 256, 0, stream>>>(agg1, (long)N * 128);
  k_zero<<<1024, 256, 0, stream>>>(agg2, (long)N * 64);
  k_zero<<<4, 256, 0, stream>>>(st, 896);

  // pack all weight matrices into WMMA B-fragment order (bf16)
  auto pack = [&](const float* s, int Ksrc, int Nc, __bf16* d, int KT, int NT) {
    int total = KT * NT * 512;
    k_pack_b<<<cdiv(total, 256), 256, 0, stream>>>(s, Ksrc, Nc, d, KT, NT);
  };
  pack(F(C1 + L_LIN1W), 128, 128, lin1p1, 4, 8);
  pack(F(C2 + L_LIN1W), 128, 64, lin1p2, 4, 4);
  pack(F(C1 + L_LIN2W), 128, 128, lin2p1, 4, 8);
  pack(F(C2 + L_LIN2W), 64, 128, lin2p2, 2, 8);
  pack(F(C1 + L_MLPW1), 50, 128, w1p1, 2, 8);
  pack(F(C2 + L_MLPW1), 50, 64, w1p2, 2, 4);
  pack(F(C1 + L_MLPW2), 128, 128, w2p1, 4, 8);
  pack(F(C2 + L_MLPW2), 64, 64, w2p2, 2, 4);
  pack(F(I_LINW), 256, 128, linp, 8, 8);
  pack(F(I_ATTNW1), 128, 64, attn1p, 4, 4);

  // per-edge scalar combined weight (both convs)
  k_combined<<<cdiv(E, 256), 256, 0, stream>>>(elen, F(C1 + L_DWW1), F(C1 + L_DWB1),
                                               F(C1 + L_DWW2), F(C1 + L_DWB2), comb1, E);
  k_combined<<<cdiv(E, 256), 256, 0, stream>>>(elen, F(C2 + L_DWW1), F(C2 + L_DWB1),
                                               F(C2 + L_DWW2), F(C2 + L_DWB2), comb2, E);

  // node transform GEMM1 (pre-BN), both convs
  int gemmGrid = cdiv((long)MT_N * 32, 256);
  k_gemm<128, 128><<<gemmGrid, 256, 0, stream>>>(x, lin1p1, F(C1 + L_LIN1B), h1, MT_N);
  k_gemm<128, 64><<<gemmGrid, 256, 0, stream>>>(x, lin1p2, F(C2 + L_LIN1B), h2, MT_N);

  // BN1 stats + fold + apply (leaky_relu 0.2), in place
  k_colstats<<<cdiv(N, 256), 128, 0, stream>>>(h1, 128, N, 256, sum_h1, sq_h1);
  k_colstats<<<cdiv(N, 256), 64, 0, stream>>>(h2, 64, N, 256, sum_h2, sq_h2);
  k_bn_finalize<<<1, 128, 0, stream>>>(sum_h1, sq_h1, F(C1 + L_BN1G), F(C1 + L_BN1B),
                                       invN, 128, scale_h1, shift_h1);
  k_bn_finalize<<<1, 128, 0, stream>>>(sum_h2, sq_h2, F(C2 + L_BN1G), F(C2 + L_BN1B),
                                       invN, 64, scale_h2, shift_h2);
  k_bn_leaky<<<cdiv((long)N * 128, 256), 256, 0, stream>>>(h1, scale_h1, shift_h1, 128,
                                                           (long)N * 128);
  k_bn_leaky<<<cdiv((long)N * 64, 256), 256, 0, stream>>>(h2, scale_h2, shift_h2, 64,
                                                          (long)N * 64);

  // fused edge pipeline: filter MLP (WMMA) -> gather -> scatter-add, both convs
  k_edge_conv<<<ET / 4, 128, 0, stream>>>(
      eattr, src, dst, comb1, comb2, h1, h2, agg1, agg2, w1p1, F(C1 + L_MLPB1),
      F(C1 + L_MLPBETA), w2p1, F(C1 + L_MLPB2), w1p2, F(C2 + L_MLPB1),
      F(C2 + L_MLPBETA), w2p2, F(C2 + L_MLPB2), ET);

  // GEMM2 per pathway (pre-BN2)
  k_gemm<128, 128><<<gemmGrid, 256, 0, stream>>>(agg1, lin2p1, F(C1 + L_LIN2B), o1, MT_N);
  k_gemm<64, 128><<<gemmGrid, 256, 0, stream>>>(agg2, lin2p2, F(C2 + L_LIN2B), o2, MT_N);

  // BN2 stats + fold
  k_colstats<<<cdiv(N, 256), 128, 0, stream>>>(o1, 128, N, 256, sum_o1, sq_o1);
  k_colstats<<<cdiv(N, 256), 128, 0, stream>>>(o2, 128, N, 256, sum_o2, sq_o2);
  k_bn_finalize<<<1, 128, 0, stream>>>(sum_o1, sq_o1, F(C1 + L_BN2G), F(C1 + L_BN2B),
                                       invN, 128, scale_o1, shift_o1);
  k_bn_finalize<<<1, 128, 0, stream>>>(sum_o2, sq_o2, F(C2 + L_BN2G), F(C2 + L_BN2B),
                                       invN, 128, scale_o2, shift_o2);

  // BN2-apply + SSP into concat buffer, then final linear
  k_concat_ssp<<<cdiv((long)N * 256, 256), 256, 0, stream>>>(
      o1, o2, scale_o1, shift_o1, scale_o2, shift_o2, F(I_ACTBETA), xcat,
      (long)N * 256);
  k_gemm<256, 128><<<gemmGrid, 256, 0, stream>>>(xcat, linp, F(I_LINB), xc, MT_N);

  // attention head
  k_gemm<128, 64><<<gemmGrid, 256, 0, stream>>>(xc, attn1p, F(I_ATTNB1), tbuf, MT_N);
  k_attn<<<cdiv(N, 128), 128, 0, stream>>>(tbuf, F(I_ATTNW2), F(I_ATTNB2), attnv, N);
  k_final<<<cdiv((long)N * 128, 256), 256, 0, stream>>>(xc, attnv, (float*)d_out,
                                                        (long)N * 128);
}
